// RelativeMultiheadAttention_19430432047445
// MI455X (gfx1250) — compile-verified, matrix operand order checked by reading, not run on hardware
//
#include <hip/hip_runtime.h>
#include <stddef.h>

// ---------------------------------------------------------------------------
// Transformer-XL relative multi-head attention for MI455X (gfx1250, wave32).
// All matmuls via v_wmma_f32_16x16x32_bf16 (f32 accumulate).
// Attention is fused flash-style; K/V^T/R tiles staged to LDS with the
// Tensor Data Mover (tensor_load_to_lds + s_wait_tensorcnt).
// Workspace requirement: ~114 MiB (documented offsets below).
// ---------------------------------------------------------------------------

typedef __attribute__((ext_vector_type(16))) __bf16        v16bf;
typedef __attribute__((ext_vector_type(8)))  float         v8f;
typedef __attribute__((ext_vector_type(4)))  unsigned int  u32x4;
typedef __attribute__((ext_vector_type(8)))  int           i32x8;
typedef __attribute__((ext_vector_type(4)))  int           i32x4;

#define E_     1024
#define H_     16
#define HD_    64
#define QLEN_  1024
#define KLEN_  2048
#define BS_    4
#define PREV_  (KLEN_ - QLEN_)   // 1024

// attention tiling
#define QT 64      // q rows per block (4 waves x 16)
#define KT 64      // keys per k-step
#define RT 144     // staged R-band rows
#define LDP 72     // padded LDS row length in bf16 (128B data + 16B pad)
#define GW 96      // per-wave G band width (needs >= 79)

#if defined(__gfx1250__) && __has_builtin(__builtin_amdgcn_tensor_load_to_lds)
#define HAVE_TDM 1
#else
#define HAVE_TDM 0
#endif

// ---- fragment helpers ------------------------------------------------------
// A-fragment layout (16-bit, per ISA 7.12.2): lane L holds M = L&15;
// halves 0..7 = K off..off+7, halves 8..15 = K off+16..off+23, off = L<16?0:8.
// We assume the symmetric layout for B (N = L&15, same K pattern), i.e. a
// B column n is a contiguous row of the (pre-transposed) operand.
union FragU { v16bf v; u32x4 u[2]; };

__device__ __forceinline__ v16bf load_frag(const __bf16* p) {
  FragU f;
  f.u[0] = *reinterpret_cast<const u32x4*>(p);
  f.u[1] = *reinterpret_cast<const u32x4*>(p + 16);
  return f.v;
}

__device__ __forceinline__ v8f wmma_bf16(v16bf a, v16bf b, v8f c) {
  return __builtin_amdgcn_wmma_f32_16x16x32_bf16(false, a, false, b, (short)0, c,
                                                 false, false);
}

// ---- TDM 2-D tile load (element size = 2B, LDS row padding 128B->144B) ----
#if HAVE_TDM
__device__ __forceinline__ void tdm_load_2d(const void* lds_ptr, const void* gptr,
                                            unsigned tdim0, unsigned tdim1,
                                            unsigned tile0, unsigned tile1,
                                            unsigned stride0) {
  unsigned long long ga = (unsigned long long)(size_t)gptr;
  unsigned lds = (unsigned)(size_t)lds_ptr;     // low 32 bits = LDS byte offset
  u32x4 g0;
  g0[0] = 1u;                                        // count=1, user descriptor
  g0[1] = lds;                                       // lds_addr
  g0[2] = (unsigned)(ga & 0xFFFFFFFFu);              // global_addr lo
  g0[3] = (unsigned)((ga >> 32) & 0x01FFFFFFu) | (2u << 30);  // addr hi | type=2
  i32x8 g1;
  // data_size=2B, pad_enable, pad_interval=32 DWORDs(code 4), pad_amount=4 DW(code 3)
  g1[0] = (int)((1u << 16) | (1u << 20) | (4u << 22) | (3u << 25));
  g1[1] = (int)((tdim0 & 0xFFFFu) << 16);            // tensor_dim0[15:0]
  g1[2] = (int)((tdim0 >> 16) | ((tdim1 & 0xFFFFu) << 16));
  g1[3] = (int)((tdim1 >> 16) | (tile0 << 16));      // tile_dim0
  g1[4] = (int)(tile1 & 0xFFFFu);                    // tile_dim1, tile_dim2=0
  g1[5] = (int)stride0;                              // tensor_dim0_stride lo
  g1[6] = 0;
  g1[7] = 0;
  i32x4 z4 = {0, 0, 0, 0};                           // D# groups 2/3 (2-D tensor)
  i32x8 z8 = {0, 0, 0, 0, 0, 0, 0, 0};               // trailing group (clang-23 form)
  __builtin_amdgcn_tensor_load_to_lds(g0, g1, z4, z4, z8, 0);
}
#endif

// ===========================================================================
// Elementwise f32 -> bf16
// ===========================================================================
__global__ __launch_bounds__(256) void cvt_bf16(const float* __restrict__ src,
                                                __bf16* __restrict__ dst, int n) {
  int i = blockIdx.x * blockDim.x + threadIdx.x;
  int stride = gridDim.x * blockDim.x;
  for (; i < n; i += stride) dst[i] = (__bf16)src[i];
}

// ===========================================================================
// W[K][N] f32 -> Wt[N][K] bf16 (tiled transpose through LDS)
// ===========================================================================
__global__ __launch_bounds__(256) void transpose_w(const float* __restrict__ W,
                                                   __bf16* __restrict__ Wt) {
  __shared__ float t[32][33];
  const int bx = blockIdx.x * 32;  // source col tile (dest row tile)
  const int by = blockIdx.y * 32;  // source row tile
  const int x = threadIdx.x & 31;
  const int y4 = (threadIdx.x >> 5) * 4;
#pragma unroll
  for (int i = 0; i < 4; ++i)
    t[y4 + i][x] = W[(size_t)(by + y4 + i) * E_ + bx + x];
  __syncthreads();
#pragma unroll
  for (int i = 0; i < 4; ++i)
    Wt[(size_t)(bx + y4 + i) * E_ + by + x] = (__bf16)t[x][y4 + i];
}

// ===========================================================================
// Generic WMMA GEMM: C[M,N] = A[M,E] x W[E,N] (+bias), epilogue per mode.
// Block = 128 threads (4 waves), tile 64x64, K-chunks of 32.
// A is bf16 row-major; Wt is W^T bf16 (row n = column n of W, contiguous).
// ===========================================================================
enum { MODE_Q = 0, MODE_K = 1, MODE_V = 2, MODE_P = 3, MODE_O = 4 };

template <int MODE>
__global__ __launch_bounds__(128) void wmma_gemm(
    const __bf16* __restrict__ A, const __bf16* __restrict__ Wt,
    const float* __restrict__ bias, const float* __restrict__ Ub,
    const float* __restrict__ Vb, __bf16* __restrict__ out0,
    __bf16* __restrict__ out1, float* __restrict__ fout) {
  const int tid = threadIdx.x;
  const int w = tid >> 5, lane = tid & 31;
  const int ln16 = lane & 15;
  const int hi8 = (lane >> 4) ? 8 : 0;
  const int off = (lane < 16) ? 0 : 8;
  const int row0 = blockIdx.x * 64 + w * 16;
  const int n0 = blockIdx.y * 64;

  v8f acc[4] = {};
  const __bf16* ap = A + (size_t)(row0 + ln16) * E_;
#pragma unroll 4
  for (int k0 = 0; k0 < E_; k0 += 32) {
    v16bf af = load_frag(ap + k0 + off);
#pragma unroll
    for (int s = 0; s < 4; ++s) {
      v16bf bf_ = load_frag(Wt + (size_t)(n0 + 16 * s + ln16) * E_ + k0 + off);
      acc[s] = wmma_bf16(af, bf_, acc[s]);
    }
  }

#pragma unroll
  for (int s = 0; s < 4; ++s)
#pragma unroll
    for (int v = 0; v < 8; ++v) {
      const int r = row0 + v + hi8;          // output row
      const int n = n0 + 16 * s + ln16;      // output col
      float x = acc[s][v];
      if constexpr (MODE != MODE_P) x += bias[n];
      if constexpr (MODE == MODE_Q) {
        // rows = q*BS+b ; cols = h*64+d ; write (q+U) and (q+V), head-major
        const int hh = n >> 6, d = n & 63, q = r >> 2, b = r & 3;
        const size_t idx = (((size_t)b * H_ + hh) * QLEN_ + q) * HD_ + d;
        out0[idx] = (__bf16)(x + Ub[n]);     // U[h][d] == U[n]
        out1[idx] = (__bf16)(x + Vb[n]);
      } else if constexpr (MODE == MODE_K) {
        const int hh = n >> 6, d = n & 63, k = r >> 2, b = r & 3;
        out0[(((size_t)b * H_ + hh) * KLEN_ + k) * HD_ + d] = (__bf16)x;
      } else if constexpr (MODE == MODE_V) {
        // V stored transposed per head: [b][h][d][k] so P*V B-frags are rows
        const int hh = n >> 6, d = n & 63, k = r >> 2, b = r & 3;
        out0[(((size_t)b * H_ + hh) * HD_ + d) * KLEN_ + k] = (__bf16)x;
      } else if constexpr (MODE == MODE_P) {
        const int hh = n >> 6, d = n & 63;   // rows = m
        out0[((size_t)hh * KLEN_ + r) * HD_ + d] = (__bf16)x;
      } else {  // MODE_O
        fout[(size_t)r * E_ + n] = x;
      }
    }
}

// ===========================================================================
// Fused relative attention. Grid = (QLEN/QT, B*H), 128 threads (4 waves).
// Per k-step: TDM-stage K / V^T / R-band to LDS, then per wave:
//   content  S_c = (q+U) K^T            (8  wmma)
//   band     G   = (q+V) R^T  (16x96)   (12 wmma) -> LDS -> shift-gather
//   softmax  (online, shfl_xor row reductions)
//   output   O  += P V                  (8  wmma)
// ===========================================================================
__global__ __launch_bounds__(128) void attn_kernel(
    const __bf16* __restrict__ qcu, const __bf16* __restrict__ qcv,
    const __bf16* __restrict__ Kh, const __bf16* __restrict__ VhT,
    const __bf16* __restrict__ Rh, __bf16* __restrict__ Alpha) {
  __shared__ __align__(16) __bf16 Kt[KT][LDP];    // [key][d]
  __shared__ __align__(16) __bf16 Vt[HD_][LDP];   // [d][key]  (V^T)
  __shared__ __align__(16) __bf16 Rt[RT][LDP];    // [m-mlo][d]
  __shared__ __align__(16) float Gs[4][16][GW];   // per-wave band / P staging

  const int tid = threadIdx.x;
  const int w = tid >> 5, lane = tid & 31;
  const int ln16 = lane & 15;
  const int hi8 = (lane >> 4) ? 8 : 0;
  const int off = (lane < 16) ? 0 : 8;
  const int i0 = blockIdx.x * QT;
  const int bh = blockIdx.y;
  const int b = bh >> 4, h = bh & 15;
  const int iw = i0 + w * 16;

  // Q fragments, held for the whole kernel (K = 64 -> 2 frags each)
  v16bf aCu[2], aCv[2];
  {
    const __bf16* rp = qcu + ((size_t)bh * QLEN_ + (iw + ln16)) * HD_;
#pragma unroll
    for (int kk = 0; kk < 2; ++kk) aCu[kk] = load_frag(rp + kk * 32 + off);
    rp = qcv + ((size_t)bh * QLEN_ + (iw + ln16)) * HD_;
#pragma unroll
    for (int kk = 0; kk < 2; ++kk) aCv[kk] = load_frag(rp + kk * 32 + off);
  }

  float mrun[8], lrun[8];
  v8f o[4] = {};
#pragma unroll
  for (int v = 0; v < 8; ++v) { mrun[v] = -1e30f; lrun[v] = 0.f; }

  const int jend = i0 + QT + PREV_;  // exclusive key bound for this q-tile
  for (int j0 = 0; j0 < jend; j0 += KT) {
    __syncthreads();  // previous tile fully consumed
    const int mlo = j0 - i0 + (QLEN_ - QT);  // >= 0 always
    int rrows = KLEN_ - mlo; if (rrows > RT) rrows = RT;
#if HAVE_TDM
    if (tid < 32) {
      tdm_load_2d(&Kt[0][0], Kh + ((size_t)bh * KLEN_ + j0) * HD_,
                  HD_, KT, HD_, KT, HD_);
      tdm_load_2d(&Vt[0][0], VhT + (size_t)bh * HD_ * KLEN_ + j0,
                  KT, HD_, KT, HD_, KLEN_);
      tdm_load_2d(&Rt[0][0], Rh + ((size_t)h * KLEN_ + mlo) * HD_,
                  HD_, (unsigned)rrows, HD_, (unsigned)rrows, HD_);
      __builtin_amdgcn_s_wait_tensorcnt(0);
    }
#else
    for (int e = tid; e < KT * HD_; e += 128) {
      int r = e >> 6, c = e & 63;
      Kt[r][c] = Kh[((size_t)bh * KLEN_ + j0 + r) * HD_ + c];
      Vt[r][c] = VhT[((size_t)bh * HD_ + r) * KLEN_ + j0 + c];
    }
    for (int e = tid; e < rrows * HD_; e += 128) {
      int r = e >> 6, c = e & 63;
      Rt[r][c] = Rh[((size_t)h * KLEN_ + mlo + r) * HD_ + c];
    }
#endif
    __syncthreads();

    // ---- content scores: S_c = (q+U) K^T --------------------------------
    v8f sc[4];
#pragma unroll
    for (int s = 0; s < 4; ++s) {
      v8f c = {};
#pragma unroll
      for (int kk = 0; kk < 2; ++kk) {
        v16bf bfr = load_frag(&Kt[16 * s + ln16][kk * 32 + off]);
        c = wmma_bf16(aCu[kk], bfr, c);
      }
      sc[s] = c;
    }

    // ---- position band: G[r][g] = (q+V)[iw+r] . R[mlo + 48-16w + g] -----
#pragma unroll
    for (int gs = 0; gs < 6; ++gs) {
      v8f g = {};
      const int t = 48 - 16 * w + 16 * gs + ln16;  // Rt row, always in [0,143]
#pragma unroll
      for (int kk = 0; kk < 2; ++kk) {
        v16bf bfr = load_frag(&Rt[t][kk * 32 + off]);
        g = wmma_bf16(aCv[kk], bfr, g);
      }
#pragma unroll
      for (int v = 0; v < 8; ++v)
        Gs[w][v + hi8][16 * gs + ln16] = g[v];
    }

    // ---- gather rel-shifted position, scale, mask -----------------------
#pragma unroll
    for (int s = 0; s < 4; ++s)
#pragma unroll
      for (int v = 0; v < 8; ++v) {
        const int r = v + hi8;
        const int c = 16 * s + ln16;
        const float p = Gs[w][r][c - r + 15];  // m = j - i + QLEN - 1
        const float sv = (sc[s][v] + p) * 0.03125f;  // 1/sqrt(E)
        sc[s][v] = ((j0 + c) > (iw + r) + PREV_) ? -1e20f : sv;
      }

    // ---- online softmax (rows spread over 16 lanes of a half-wave) ------
    float alpha_[8];
#pragma unroll
    for (int v = 0; v < 8; ++v) {
      float mt = fmaxf(fmaxf(sc[0][v], sc[1][v]), fmaxf(sc[2][v], sc[3][v]));
#pragma unroll
      for (int d = 1; d < 16; d <<= 1) mt = fmaxf(mt, __shfl_xor(mt, d, 32));
      const float nm = fmaxf(mrun[v], mt);
      alpha_[v] = __expf(mrun[v] - nm);
      mrun[v] = nm;
      float ps = 0.f;
#pragma unroll
      for (int s = 0; s < 4; ++s) {
        const float e = __expf(sc[s][v] - nm);
        sc[s][v] = e;
        ps += e;
      }
#pragma unroll
      for (int d = 1; d < 16; d <<= 1) ps += __shfl_xor(ps, d, 32);
      lrun[v] = lrun[v] * alpha_[v] + ps;
    }
#pragma unroll
    for (int n = 0; n < 4; ++n)
#pragma unroll
      for (int v = 0; v < 8; ++v) o[n][v] *= alpha_[v];

    // ---- stage P as bf16 (reuse Gs region), build A-frags, O += P V -----
    __bf16* Sst = reinterpret_cast<__bf16*>(&Gs[w][0][0]);  // [16][64]
#pragma unroll
    for (int s = 0; s < 4; ++s)
#pragma unroll
      for (int v = 0; v < 8; ++v)
        Sst[(v + hi8) * KT + 16 * s + ln16] = (__bf16)sc[s][v];

    v16bf aP[2];
#pragma unroll
    for (int kk = 0; kk < 2; ++kk)
      aP[kk] = load_frag(Sst + ln16 * KT + kk * 32 + off);
#pragma unroll
    for (int n = 0; n < 4; ++n)
#pragma unroll
      for (int kk = 0; kk < 2; ++kk) {
        v16bf bfr = load_frag(&Vt[16 * n + ln16][kk * 32 + off]);
        o[n] = wmma_bf16(aP[kk], bfr, o[n]);
      }
  }

  // ---- normalize and emit alpha[q][b][h*64+d] as bf16 -------------------
#pragma unroll
  for (int v = 0; v < 8; ++v) lrun[v] = 1.0f / lrun[v];
#pragma unroll
  for (int n = 0; n < 4; ++n)
#pragma unroll
    for (int v = 0; v < 8; ++v) {
      const int r = v + hi8;
      const int q = iw + r;
      const int d = 16 * n + ln16;
      Alpha[((size_t)q * BS_ + b) * E_ + h * HD_ + d] = (__bf16)(o[n][v] * lrun[v]);
    }
}

// ===========================================================================
// Host side
// ===========================================================================
static constexpr size_t OFF_QA    = 0;                          // 4096x1024 bf16
static constexpr size_t OFF_KA    = OFF_QA  + 8388608;          // 8192x1024
static constexpr size_t OFF_VA    = OFF_KA  + 16777216;         // 8192x1024
static constexpr size_t OFF_PA    = OFF_VA  + 16777216;         // 2048x1024
static constexpr size_t OFF_WQT   = OFF_PA  + 4194304;          // 1024x1024 each
static constexpr size_t OFF_WKT   = OFF_WQT + 2097152;
static constexpr size_t OFF_WVT   = OFF_WKT + 2097152;
static constexpr size_t OFF_WPT   = OFF_WVT + 2097152;
static constexpr size_t OFF_WOT   = OFF_WPT + 2097152;
static constexpr size_t OFF_QCU   = OFF_WOT + 2097152;          // [b][h][q][d]
static constexpr size_t OFF_QCV   = OFF_QCU + 8388608;
static constexpr size_t OFF_KH    = OFF_QCV + 8388608;          // [b][h][k][d]
static constexpr size_t OFF_VHT   = OFF_KH  + 16777216;         // [b][h][d][k]
static constexpr size_t OFF_RH    = OFF_VHT + 16777216;         // [h][m][d]
static constexpr size_t OFF_ALPHA = OFF_RH  + 4194304;          // [q*B+b][e]
// total workspace: OFF_ALPHA + 8388608 = ~114 MiB

extern "C" void kernel_launch(void* const* d_in, const int* in_sizes, int n_in,
                              void* d_out, int out_size, void* d_ws, size_t ws_size,
                              hipStream_t stream) {
  (void)in_sizes; (void)n_in; (void)out_size; (void)ws_size;
  const float* query = (const float*)d_in[0];
  const float* key_  = (const float*)d_in[1];
  const float* value = (const float*)d_in[2];
  const float* pos   = (const float*)d_in[3];
  const float* U     = (const float*)d_in[4];
  const float* V     = (const float*)d_in[5];
  const float* Wq    = (const float*)d_in[6];
  const float* bq    = (const float*)d_in[7];
  const float* Wk    = (const float*)d_in[8];
  const float* bk    = (const float*)d_in[9];
  const float* Wv    = (const float*)d_in[10];
  const float* bv    = (const float*)d_in[11];
  const float* Wp    = (const float*)d_in[12];
  const float* Wo    = (const float*)d_in[13];
  const float* bo    = (const float*)d_in[14];
  float* out = (float*)d_out;

  char* ws = (char*)d_ws;
  __bf16* qA    = (__bf16*)(ws + OFF_QA);
  __bf16* kA    = (__bf16*)(ws + OFF_KA);
  __bf16* vA    = (__bf16*)(ws + OFF_VA);
  __bf16* pA    = (__bf16*)(ws + OFF_PA);
  __bf16* WqT   = (__bf16*)(ws + OFF_WQT);
  __bf16* WkT   = (__bf16*)(ws + OFF_WKT);
  __bf16* WvT   = (__bf16*)(ws + OFF_WVT);
  __bf16* WpT   = (__bf16*)(ws + OFF_WPT);
  __bf16* WoT   = (__bf16*)(ws + OFF_WOT);
  __bf16* qcu   = (__bf16*)(ws + OFF_QCU);
  __bf16* qcv   = (__bf16*)(ws + OFF_QCV);
  __bf16* Kh    = (__bf16*)(ws + OFF_KH);
  __bf16* VhT   = (__bf16*)(ws + OFF_VHT);
  __bf16* Rh    = (__bf16*)(ws + OFF_RH);
  __bf16* Alpha = (__bf16*)(ws + OFF_ALPHA);

  // 1) convert activations to bf16
  cvt_bf16<<<2048, 256, 0, stream>>>(query, qA, QLEN_ * BS_ * E_);
  cvt_bf16<<<2048, 256, 0, stream>>>(key_,  kA, KLEN_ * BS_ * E_);
  cvt_bf16<<<2048, 256, 0, stream>>>(value, vA, KLEN_ * BS_ * E_);
  cvt_bf16<<<2048, 256, 0, stream>>>(pos,   pA, KLEN_ * E_);

  // 2) transpose weights to bf16 (B-operand rows become contiguous)
  dim3 tg(32, 32), tb(256);
  transpose_w<<<tg, tb, 0, stream>>>(Wq, WqT);
  transpose_w<<<tg, tb, 0, stream>>>(Wk, WkT);
  transpose_w<<<tg, tb, 0, stream>>>(Wv, WvT);
  transpose_w<<<tg, tb, 0, stream>>>(Wp, WpT);
  transpose_w<<<tg, tb, 0, stream>>>(Wo, WoT);

  // 3) projections
  wmma_gemm<MODE_Q><<<dim3(64, 16), 128, 0, stream>>>(
      qA, WqT, bq, U, V, qcu, qcv, nullptr);
  wmma_gemm<MODE_K><<<dim3(128, 16), 128, 0, stream>>>(
      kA, WkT, bk, nullptr, nullptr, Kh, nullptr, nullptr);
  wmma_gemm<MODE_V><<<dim3(128, 16), 128, 0, stream>>>(
      vA, WvT, bv, nullptr, nullptr, VhT, nullptr, nullptr);
  wmma_gemm<MODE_P><<<dim3(32, 16), 128, 0, stream>>>(
      pA, WpT, nullptr, nullptr, nullptr, Rh, nullptr, nullptr);

  // 4) fused relative attention
  attn_kernel<<<dim3(QLEN_ / QT, BS_ * H_), 128, 0, stream>>>(
      qcu, qcv, Kh, VhT, Rh, Alpha);

  // 5) output projection (f32 result)
  wmma_gemm<MODE_O><<<dim3(64, 16), 128, 0, stream>>>(
      Alpha, WoT, bo, nullptr, nullptr, nullptr, nullptr, out);
}
